// MultiHeadDotProductAttention_53051436040499
// MI455X (gfx1250) — matrix operand order repaired and checked by reading, not compiled, not run on hardware
//
#include <hip/hip_runtime.h>
#include <hip/hip_bf16.h>

// ---------------------------------------------------------------------------
// Types (trivial ext-vector types so unions/arrays stay trivially constructible)
// ---------------------------------------------------------------------------
typedef __bf16 bf16_t;
typedef __attribute__((ext_vector_type(16))) __bf16 bf16x16;
typedef __attribute__((ext_vector_type(8)))  float  v8f;
typedef __attribute__((ext_vector_type(4)))  float  f32x4;
typedef __attribute__((ext_vector_type(4)))  unsigned int u32x4;
typedef __attribute__((ext_vector_type(8)))  int    i32x8;
typedef __attribute__((ext_vector_type(4)))  int    i32x4;

union AFrag { bf16x16 v; f32x4 q[2]; };

static __device__ inline v8f vzero8() {
  v8f z = {0.f, 0.f, 0.f, 0.f, 0.f, 0.f, 0.f, 0.f};
  return z;
}

// Problem constants
#define BATCH 2
#define SEQ   2048
#define HEADS 16
#define DH    64
#define DMODEL 1024
#define MROWS (BATCH * SEQ) // 4096

// ---------------------------------------------------------------------------
// TDM: one-instruction DMA of a 2D bf16 tile (tileW x tileH elements, row
// stride rowStride elements) from global memory into LDS at byte offset
// ldsOff, inserting 16B of padding after every 128B row (pad_interval=4 ->
// 2^4*8B = 128B trigger, pad_amount=3 -> 4 DWORDs) so the LDS image matches
// the [rows][72] bf16 padded tiles used by the WMMA fragment readers.
// D# packing per CDNA5 ISA.
// ---------------------------------------------------------------------------
static __device__ inline void tdm_load_2d_bf16(unsigned ldsOff,
                                               const bf16_t* gptr,
                                               unsigned tileW, unsigned tileH,
                                               unsigned rowStride) {
  unsigned long long ga = (unsigned long long)(uintptr_t)gptr;
  u32x4 g0;
  g0.x = 1u;                                   // count=1, user descriptor
  g0.y = ldsOff;                               // lds_addr (bytes)
  g0.z = (unsigned)(ga & 0xFFFFFFFFu);         // global_addr[31:0]
  g0.w = (unsigned)((ga >> 32) & 0x01FFFFFFu)  // global_addr[56:32]
         | (2u << 30);                         // type = 2 ("image")
  i32x8 g1;
  g1[0] = (int)((1u << 16)        // data_size = 1 -> 2 bytes/elem
                | (1u << 20)      // pad_enable
                | (4u << 22)      // pad_interval: pad after 128B
                | (3u << 25));    // pad_amount: 4 DWORDs = 16B
  g1[1] = (int)((tileW & 0xFFFFu) << 16);                       // tensor_dim0 lo16
  g1[2] = (int)(((tileW >> 16) & 0xFFFFu) |
                ((tileH & 0xFFFFu) << 16));                     // dim0 hi | dim1 lo
  g1[3] = (int)(((tileH >> 16) & 0xFFFFu) |
                ((tileW & 0xFFFFu) << 16));                     // dim1 hi | tile_dim0
  g1[4] = (int)(tileH & 0xFFFFu);                               // tile_dim1 (dim2=0)
  g1[5] = (int)rowStride;                                       // dim0_stride lo32
  g1[6] = 0;                                                    // stride hi bits
  g1[7] = 0;
  i32x4 z4 = {0, 0, 0, 0};
#if __clang_major__ >= 23
  i32x8 z8 = {0, 0, 0, 0, 0, 0, 0, 0};
  __builtin_amdgcn_tensor_load_to_lds(g0, g1, z4, z4, z8, 0);
#else
  __builtin_amdgcn_tensor_load_to_lds(g0, g1, z4, z4, 0);
#endif
}

// ---------------------------------------------------------------------------
// Kernel 0: fp32 -> bf16 conversion (one element per thread)
// ---------------------------------------------------------------------------
__global__ __launch_bounds__(256) void cvt_f32_bf16(const float* __restrict__ in,
                                                    bf16_t* __restrict__ out, int n) {
  int i = blockIdx.x * 256 + threadIdx.x;
  if (i < n) out[i] = (bf16_t)in[i];
}

// ---------------------------------------------------------------------------
// Kernel 1: NT GEMM  C[M,N] = A[M,K] * W[N,K]^T   (K = N = 1024, M = 4096)
// mode 0: store bf16 as Q layout [B,H,S,Dh]
// mode 1: store bf16 as K layout [B,H,S,Dh]
// mode 2: store bf16 as Vt layout [B,H,Dh,S]   (transposed V)
// mode 3: store fp32 row-major [M,N] (final output projection)
// ---------------------------------------------------------------------------
__global__ __launch_bounds__(256) void gemm_nt_bf16(const bf16_t* __restrict__ A,
                                                    const bf16_t* __restrict__ W,
                                                    bf16_t* __restrict__ outb,
                                                    float* __restrict__ outf,
                                                    int mode) {
  __shared__ __align__(16) bf16_t lA[128][40]; // 32 + 8 pad -> 80B pitch
  __shared__ __align__(16) bf16_t lB[128][40];

  const int tid  = threadIdx.x;
  const int lane = tid & 31;
  const int w    = tid >> 5;      // wave id 0..7
  const int hl   = lane >> 4;     // half-wave select
  const int ln   = lane & 15;

  const int m0 = blockIdx.x * 128;
  const int n0 = blockIdx.y * 128;
  const int wm = (w & 3) * 32;    // wave M offset (4 waves in M)
  const int wn = (w >> 2) * 64;   // wave N offset (2 waves in N)

  v8f acc[2][4];
#pragma unroll
  for (int i = 0; i < 2; i++)
#pragma unroll
    for (int j = 0; j < 4; j++) acc[i][j] = vzero8();

  for (int k0 = 0; k0 < DMODEL; k0 += 32) {
    __syncthreads();
    // Cooperative coalesced staging: 512 x 16B chunks per tile, 2 per thread.
#pragma unroll
    for (int i = 0; i < 2; i++) {
      int c = tid + i * 256;          // 0..511
      int row = c >> 2, seg = c & 3;  // 4 x 16B per 64B row
      const f32x4* gA = (const f32x4*)(A + (size_t)(m0 + row) * DMODEL + k0) + seg;
      *(f32x4*)(&lA[row][seg * 8]) = *gA;
      const f32x4* gB = (const f32x4*)(W + (size_t)(n0 + row) * DMODEL + k0) + seg;
      *(f32x4*)(&lB[row][seg * 8]) = *gB;
    }
    __syncthreads();

    // A fragments: lane(m=ln, half=hl) holds K[8h..8h+7] and K[16+8h..16+8h+7]
    AFrag a[2], b[4];
#pragma unroll
    for (int mi = 0; mi < 2; mi++) {
      const bf16_t* p = &lA[wm + mi * 16 + ln][hl * 8];
      a[mi].q[0] = *(const f32x4*)p;
      a[mi].q[1] = *(const f32x4*)(p + 16);
    }
    // B fragments: lane(n=ln, half=hl) holds K[8h..8h+7] and K[16+8h..16+8h+7]
#pragma unroll
    for (int ni = 0; ni < 4; ni++) {
      const bf16_t* p = &lB[wn + ni * 16 + ln][hl * 8];
      b[ni].q[0] = *(const f32x4*)p;
      b[ni].q[1] = *(const f32x4*)(p + 16);
    }
#pragma unroll
    for (int mi = 0; mi < 2; mi++)
#pragma unroll
      for (int ni = 0; ni < 4; ni++)
        acc[mi][ni] = __builtin_amdgcn_wmma_f32_16x16x32_bf16(
            false, a[mi].v, false, b[ni].v, (short)0, acc[mi][ni], false, false);
  }

  // Epilogue. C layout: lane ln holds N=ln; VGPR r holds M = r + 8*hl.
#pragma unroll
  for (int mi = 0; mi < 2; mi++)
#pragma unroll
    for (int ni = 0; ni < 4; ni++) {
      int ng = n0 + wn + ni * 16 + ln;
#pragma unroll
      for (int r = 0; r < 8; r++) {
        int mg = m0 + wm + mi * 16 + r + 8 * hl;
        float val = acc[mi][ni][r];
        if (mode == 3) {
          outf[(size_t)mg * DMODEL + ng] = val;
        } else {
          int bb = mg >> 11, ss = mg & (SEQ - 1);
          int hh = ng >> 6,  dd = ng & (DH - 1);
          if (mode == 2)  // V transposed: [B,H,Dh,S]
            outb[(((size_t)bb * HEADS + hh) * DH + dd) * SEQ + ss] = (bf16_t)val;
          else            // Q/K: [B,H,S,Dh]
            outb[(((size_t)bb * HEADS + hh) * SEQ + ss) * DH + dd] = (bf16_t)val;
        }
      }
    }
}

// ---------------------------------------------------------------------------
// Kernel 2: causal flash attention.
// Q,K: [B*H, S, 64] bf16.  Vt: [B*H, 64, S] bf16.  O: [B, S, 1024] bf16.
// Grid: (S/128, B*H). 256 threads = 8 waves; wave w owns rows q0+16w..+15.
// Tile staging (Q / K / Vt) is done by the Tensor Data Mover: one
// tensor_load_to_lds per tile issued by wave 0, completion via TENSORcnt.
// ---------------------------------------------------------------------------
__global__ __launch_bounds__(256) void flash_attn_causal(const bf16_t* __restrict__ Q,
                                                         const bf16_t* __restrict__ K,
                                                         const bf16_t* __restrict__ Vt,
                                                         bf16_t* __restrict__ O) {
  __shared__ __align__(16) bf16_t lQ[128][72];   // 64 + 8 pad -> 144B pitch
  __shared__ __align__(16) bf16_t lK[64][72];
  __shared__ __align__(16) bf16_t lV[64][72];
  __shared__ __align__(16) bf16_t lP[8][16][72]; // per-wave P restage

  const int tid  = threadIdx.x;
  const int lane = tid & 31;
  const int w    = tid >> 5;
  const int hl   = lane >> 4;
  const int ln   = lane & 15;

  const int qt = blockIdx.x;
  const int bh = blockIdx.y;
  const int q0 = qt * 128;

  const bf16_t* Qh = Q  + (size_t)bh * SEQ * DH;
  const bf16_t* Kh = K  + (size_t)bh * SEQ * DH;
  const bf16_t* Vh = Vt + (size_t)bh * DH * SEQ;

  const unsigned ldsQ = (unsigned)(uintptr_t)(&lQ[0][0]);
  const unsigned ldsK = (unsigned)(uintptr_t)(&lK[0][0]);
  const unsigned ldsV = (unsigned)(uintptr_t)(&lV[0][0]);

  // Stage Q tile (128 x 64) via TDM.
  if (w == 0) {
    tdm_load_2d_bf16(ldsQ, Qh + (size_t)q0 * DH, DH, 128, DH);
    __builtin_amdgcn_s_wait_tensorcnt(0);
  }
  __syncthreads();

  // Q A-fragments for K-dim halves [0,32) and [32,64)
  AFrag qa[2];
  {
    const bf16_t* p = &lQ[w * 16 + ln][0];
    qa[0].q[0] = *(const f32x4*)(p + hl * 8);
    qa[0].q[1] = *(const f32x4*)(p + 16 + hl * 8);
    qa[1].q[0] = *(const f32x4*)(p + 32 + hl * 8);
    qa[1].q[1] = *(const f32x4*)(p + 48 + hl * 8);
  }

  float mrow[8], lrow[8];
  v8f oacc[4];
#pragma unroll
  for (int r = 0; r < 8; r++) { mrow[r] = -1e30f; lrow[r] = 0.f; }
#pragma unroll
  for (int d = 0; d < 4; d++) oacc[d] = vzero8();

  const int ntiles = qt * 2 + 2;          // causal: keys up to q0+127
  const int qmax   = q0 + w * 16 + 15;    // last row this wave owns

  for (int kt = 0; kt < ntiles; kt++) {
    const int k0 = kt * 64;
    __syncthreads();
    // Stage K (64x64 rows) and Vt (64 d-rows x 64 keys, row stride S) via TDM.
    if (w == 0) {
      tdm_load_2d_bf16(ldsK, Kh + (size_t)k0 * DH, DH, 64, DH);
      tdm_load_2d_bf16(ldsV, Vh + k0, 64, DH, SEQ);
      __builtin_amdgcn_s_wait_tensorcnt(0);
    }
    __syncthreads();
    if (k0 > qmax) continue;              // wave-uniform skip of fully-masked tile

    // scores S = Q K^T : 4 N-tiles x 2 K-steps = 8 WMMAs
    v8f sc[4];
    AFrag bk;
#pragma unroll
    for (int ni = 0; ni < 4; ni++) {
      sc[ni] = vzero8();
      const bf16_t* p = &lK[ni * 16 + ln][hl * 8];
      bk.q[0] = *(const f32x4*)p;
      bk.q[1] = *(const f32x4*)(p + 16);
      sc[ni] = __builtin_amdgcn_wmma_f32_16x16x32_bf16(false, qa[0].v, false, bk.v,
                                                       (short)0, sc[ni], false, false);
      p = &lK[ni * 16 + ln][32 + hl * 8];
      bk.q[0] = *(const f32x4*)p;
      bk.q[1] = *(const f32x4*)(p + 16);
      sc[ni] = __builtin_amdgcn_wmma_f32_16x16x32_bf16(false, qa[1].v, false, bk.v,
                                                       (short)0, sc[ni], false, false);
    }

    // mask + scale + online softmax (row r lives in the 16-lane half group)
#pragma unroll
    for (int r = 0; r < 8; r++) {
      int qg = q0 + w * 16 + r + 8 * hl;
      float sv[4];
      float tmax = -1e30f;
#pragma unroll
      for (int ni = 0; ni < 4; ni++) {
        int kg = k0 + ni * 16 + ln;
        float s = sc[ni][r] * 0.125f;     // 1/sqrt(64)
        if (kg > qg) s = -1e30f;
        sv[ni] = s;
        tmax = fmaxf(tmax, s);
      }
      for (int off = 1; off < 16; off <<= 1)
        tmax = fmaxf(tmax, __shfl_xor(tmax, off, 32));
      float mnew  = fmaxf(mrow[r], tmax);
      float alpha = __expf(mrow[r] - mnew);
      mrow[r] = mnew;
      float tsum = 0.f;
#pragma unroll
      for (int ni = 0; ni < 4; ni++) {
        float pv = (sv[ni] < -1e29f) ? 0.f : __expf(sv[ni] - mnew);
        sc[ni][r] = pv;                   // reuse as P
        tsum += pv;
      }
      for (int off = 1; off < 16; off <<= 1) tsum += __shfl_xor(tsum, off, 32);
      lrow[r] = lrow[r] * alpha + tsum;
#pragma unroll
      for (int di = 0; di < 4; di++) oacc[di][r] *= alpha;
    }

    // Restage P (C layout -> A layout) through wave-private LDS slab
#pragma unroll
    for (int r = 0; r < 8; r++)
#pragma unroll
      for (int ni = 0; ni < 4; ni++)
        lP[w][r + 8 * hl][ni * 16 + ln] = (bf16_t)sc[ni][r];
    asm volatile("s_wait_dscnt 0x0" ::: "memory"); // same-wave LDS RAW fence

    AFrag pa[2];
    {
      const bf16_t* p = &lP[w][ln][0];
      pa[0].q[0] = *(const f32x4*)(p + hl * 8);
      pa[0].q[1] = *(const f32x4*)(p + 16 + hl * 8);
      pa[1].q[0] = *(const f32x4*)(p + 32 + hl * 8);
      pa[1].q[1] = *(const f32x4*)(p + 48 + hl * 8);
    }

    // O += P * V   (Vt rows are d, contraction over k) : 8 WMMAs
    AFrag bv;
#pragma unroll
    for (int di = 0; di < 4; di++) {
      const bf16_t* p = &lV[di * 16 + ln][hl * 8];
      bv.q[0] = *(const f32x4*)p;
      bv.q[1] = *(const f32x4*)(p + 16);
      oacc[di] = __builtin_amdgcn_wmma_f32_16x16x32_bf16(false, pa[0].v, false, bv.v,
                                                         (short)0, oacc[di], false, false);
      p = &lV[di * 16 + ln][32 + hl * 8];
      bv.q[0] = *(const f32x4*)p;
      bv.q[1] = *(const f32x4*)(p + 16);
      oacc[di] = __builtin_amdgcn_wmma_f32_16x16x32_bf16(false, pa[1].v, false, bv.v,
                                                         (short)0, oacc[di], false, false);
    }
  }

  // Normalize and store O as [B, S, H*Dh] bf16
  const int bb = bh >> 4, head = bh & 15;
#pragma unroll
  for (int r = 0; r < 8; r++) {
    float inv = 1.0f / lrow[r];
    int sg = q0 + w * 16 + r + 8 * hl;
#pragma unroll
    for (int di = 0; di < 4; di++) {
      float o = oacc[di][r] * inv;
      O[((size_t)bb * SEQ + sg) * DMODEL + head * DH + di * 16 + ln] = (bf16_t)o;
    }
  }
}

// ---------------------------------------------------------------------------
// Launch
// ---------------------------------------------------------------------------
extern "C" void kernel_launch(void* const* d_in, const int* in_sizes, int n_in,
                              void* d_out, int out_size, void* d_ws, size_t ws_size,
                              hipStream_t stream) {
  const float* xq  = (const float*)d_in[0];
  const float* xkv = (const float*)d_in[1];
  // d_in[2] = mask (causal tril) — implemented analytically in the kernel
  const float* Wq = (const float*)d_in[3];
  const float* Wk = (const float*)d_in[4];
  const float* Wv = (const float*)d_in[5];
  const float* Wo = (const float*)d_in[6];
  float* out = (float*)d_out;

  char* ws = (char*)d_ws;
  const size_t MB = 1u << 20;
  bf16_t* xq_b  = (bf16_t*)(ws + 0 * MB);
  bf16_t* xkv_b = (bf16_t*)(ws + 8 * MB);
  bf16_t* Wq_b  = (bf16_t*)(ws + 16 * MB);
  bf16_t* Wk_b  = (bf16_t*)(ws + 18 * MB);
  bf16_t* Wv_b  = (bf16_t*)(ws + 20 * MB);
  bf16_t* Wo_b  = (bf16_t*)(ws + 22 * MB);
  bf16_t* Qb    = (bf16_t*)(ws + 24 * MB);
  bf16_t* Kb    = (bf16_t*)(ws + 32 * MB);
  bf16_t* Vtb   = (bf16_t*)(ws + 40 * MB);
  bf16_t* Ob    = (bf16_t*)(ws + 48 * MB);

  const int nAct = MROWS * DMODEL;   // 4194304
  const int nW   = DMODEL * DMODEL;  // 1048576
  cvt_f32_bf16<<<(nAct + 255) / 256, 256, 0, stream>>>(xq,  xq_b,  nAct);
  cvt_f32_bf16<<<(nAct + 255) / 256, 256, 0, stream>>>(xkv, xkv_b, nAct);
  cvt_f32_bf16<<<(nW + 255) / 256, 256, 0, stream>>>(Wq, Wq_b, nW);
  cvt_f32_bf16<<<(nW + 255) / 256, 256, 0, stream>>>(Wk, Wk_b, nW);
  cvt_f32_bf16<<<(nW + 255) / 256, 256, 0, stream>>>(Wv, Wv_b, nW);
  cvt_f32_bf16<<<(nW + 255) / 256, 256, 0, stream>>>(Wo, Wo_b, nW);

  dim3 gGemm(MROWS / 128, DMODEL / 128);  // 32 x 8
  gemm_nt_bf16<<<gGemm, 256, 0, stream>>>(xq_b,  Wq_b, Qb,  nullptr, 0);
  gemm_nt_bf16<<<gGemm, 256, 0, stream>>>(xkv_b, Wk_b, Kb,  nullptr, 1);
  gemm_nt_bf16<<<gGemm, 256, 0, stream>>>(xkv_b, Wv_b, Vtb, nullptr, 2);

  dim3 gAttn(SEQ / 128, BATCH * HEADS);   // 16 x 32
  flash_attn_causal<<<gAttn, 256, 0, stream>>>(Qb, Kb, Vtb, Ob);

  gemm_nt_bf16<<<gGemm, 256, 0, stream>>>(Ob, Wo_b, nullptr, out, 3);
}
